// NeuroBiMambaBlock_22411139350796
// MI455X (gfx1250) — matrix operand order repaired and checked by reading, not compiled, not run on hardware
//
#include <hip/hip_runtime.h>
#include <hip/hip_bf16.h>

// ---------------------------------------------------------------------------
// NeuroBiMambaBlock for MI455X (gfx1250): f16 WMMA GEMMs + fused elementwise
// B=4, L=4096, DM=256, DH=512, N=16, K(conv)=4, R=32
// ---------------------------------------------------------------------------

typedef _Float16 half_t;
typedef __attribute__((ext_vector_type(16))) _Float16 v16h;
typedef __attribute__((ext_vector_type(8)))  float    v8f;
typedef __attribute__((ext_vector_type(4)))  unsigned int u32x4;

#define BATCH 4
#define SEQ   4096
#define DMODEL 256
#define DHID  512
#define NSTATE 16
#define RRANK 32
#define MROWS (BATCH * SEQ)      // 16384

#define BM 128
#define BN 64
#define BK 32

#define EPI_F32    0
#define EPI_GATE16 1
#define EPI_RESID  2

union AFrag { v16h v; unsigned int d[8]; };
union BFrag { v16h v; u32x4 q[2]; };

// ---------------------------------------------------------------------------
// Tiled WMMA GEMM: C[M,N] = A[M,K] (f16 row-major) x W[N,K]^T (f16 row-major)
// 8 waves per block; wave = 32x32 output (2x2 tiles of v_wmma_f32_16x16x32_f16)
// ---------------------------------------------------------------------------
__global__ __launch_bounds__(256) void k_wmma_gemm(
    const half_t* __restrict__ A, int lda,
    const half_t* __restrict__ W, int ldw,
    float* __restrict__ C, int ldc,
    half_t* __restrict__ C16, int ldc16, int colbase,
    const float* __restrict__ aux, int ldaux,
    int M, int N, int K, int epi)
{
    __shared__ half_t As[BM * BK];   // 8 KB
    __shared__ half_t Ws[BN * BK];   // 4 KB

    const int tid  = threadIdx.x;
    const int lane = tid & 31;
    const int wave = tid >> 5;       // 0..7
    const int wm   = wave >> 1;      // 0..3 : 32-row group
    const int wn   = wave & 1;       // 0..1 : 32-col group
    const int m0   = blockIdx.y * BM;
    const int n0   = blockIdx.x * BN;
    const int hi   = (lane >> 4) & 1;  // lane group (0: lanes 0-15, 1: 16-31)
    const int l16  = lane & 15;

    v8f acc[2][2];
    for (int i = 0; i < 2; ++i)
        for (int j = 0; j < 2; ++j)
            for (int r = 0; r < 8; ++r) acc[i][j][r] = 0.0f;

    for (int k0 = 0; k0 < K; k0 += BK) {
        // stage A tile: 128x32 halves, 16 contiguous halves per thread
        {
            int idx = tid * 16;
            int r   = idx >> 5;
            int kk  = idx & 31;
            const half_t* src = A + (size_t)(m0 + r) * lda + k0 + kk;
            *(u32x4*)(&As[idx])     = *(const u32x4*)(src);
            *(u32x4*)(&As[idx + 8]) = *(const u32x4*)(src + 8);
        }
        // stage W tile: 64x32 halves, 8 contiguous halves per thread
        {
            int idx = tid * 8;
            int r   = idx >> 5;
            int kk  = idx & 31;
            *(u32x4*)(&Ws[idx]) =
                *(const u32x4*)(W + (size_t)(n0 + r) * ldw + k0 + kk);
        }
        __syncthreads();

        // prefetch next A tile rows (gfx1250 global_prefetch_b8)
        if (k0 + BK < K)
            __builtin_prefetch(A + (size_t)(m0 + (tid >> 1)) * lda + k0 + BK, 0, 1);

        // A fragments: 16x32 MxK layout (lanes 0-15: K 0-7/16-23; 16-31: +8)
        AFrag af[2];
        for (int tm = 0; tm < 2; ++tm) {
            int mloc = wm * 32 + tm * 16 + l16;
            for (int j = 0; j < 8; ++j) {
                int kpair = ((j < 4) ? (2 * j) : (2 * j + 8)) + 8 * hi;
                af[tm].d[j] = *(const unsigned int*)&As[mloc * BK + kpair];
            }
        }
        // B fragments: 32x16 KxN layout (lanes 0-15: K 0-7/16-23; 16-31: +8)
        BFrag bf[2];
        for (int tn = 0; tn < 2; ++tn) {
            int nloc = wn * 32 + tn * 16 + l16;
            bf[tn].q[0] = *(const u32x4*)&Ws[nloc * BK + hi * 8];
            bf[tn].q[1] = *(const u32x4*)&Ws[nloc * BK + 16 + hi * 8];
        }

        for (int tm = 0; tm < 2; ++tm)
            for (int tn = 0; tn < 2; ++tn)
                acc[tm][tn] = __builtin_amdgcn_wmma_f32_16x16x32_f16(
                    false, af[tm].v, false, bf[tn].v,
                    (short)0, acc[tm][tn], false, false);
        __syncthreads();
    }

    // epilogue: C/D layout — lane group hi gives M = r + 8*hi, N = lane&15
    for (int tm = 0; tm < 2; ++tm) {
        for (int tn = 0; tn < 2; ++tn) {
            int colg = n0 + wn * 32 + tn * 16 + l16;
            for (int r = 0; r < 8; ++r) {
                int rowg = m0 + wm * 32 + tm * 16 + r + 8 * hi;
                float v = acc[tm][tn][r];
                if (epi == EPI_F32) {
                    C[(size_t)rowg * ldc + colg] = v;
                } else if (epi == EPI_GATE16) {
                    float g  = aux[(size_t)rowg * ldaux + colg];
                    float sg = g / (1.0f + __expf(-g));
                    C16[(size_t)rowg * ldc16 + colbase + colg] = (half_t)(v * sg);
                } else { // EPI_RESID
                    C[(size_t)rowg * ldc + colg] =
                        v + aux[(size_t)rowg * ldaux + colg];
                }
            }
        }
    }
}

// ---------------------------------------------------------------------------
// f32 -> f16 convert
// ---------------------------------------------------------------------------
__global__ void k_f32_to_f16(const float* __restrict__ src,
                             half_t* __restrict__ dst, int n)
{
    int i = blockIdx.x * blockDim.x + threadIdx.x;
    if (i < n) dst[i] = (half_t)src[i];
}

// ---------------------------------------------------------------------------
// LayerNorm over DM=256, one wave32 per row, output f16 for the first GEMM
// ---------------------------------------------------------------------------
__global__ __launch_bounds__(256) void k_layernorm(
    const float* __restrict__ x, const float* __restrict__ g,
    const float* __restrict__ bp, half_t* __restrict__ h16)
{
    int wave = threadIdx.x >> 5;
    int lane = threadIdx.x & 31;
    size_t row = (size_t)blockIdx.x * 8 + wave;
    const float* xr = x + row * DMODEL;

    float v[8], s = 0.f, ss = 0.f;
    for (int j = 0; j < 8; ++j) {
        v[j] = xr[lane + 32 * j];
        s  += v[j];
        ss += v[j] * v[j];
    }
    for (int off = 16; off > 0; off >>= 1) {
        s  += __shfl_xor(s,  off, 32);
        ss += __shfl_xor(ss, off, 32);
    }
    float mean = s * (1.0f / DMODEL);
    float var  = ss * (1.0f / DMODEL) - mean * mean;
    float rs   = rsqrtf(var + 1e-5f);
    for (int j = 0; j < 8; ++j) {
        int c = lane + 32 * j;
        h16[row * DMODEL + c] = (half_t)((v[j] - mean) * rs * g[c] + bp[c]);
    }
}

// ---------------------------------------------------------------------------
// Causal depthwise conv (K=4) + SiLU. rev=1 computes the flipped-sequence
// conv in physical order (anti-causal, mirrored taps).
// dst stride is DHID; src stride is sstride (column slice of a wider buffer).
// ---------------------------------------------------------------------------
__global__ void k_dwconv_silu(const float* __restrict__ src, int sstride,
                              const float* __restrict__ w,
                              const float* __restrict__ bias,
                              float* __restrict__ dst32,
                              half_t* __restrict__ dst16, int rev)
{
    size_t gid = (size_t)blockIdx.x * blockDim.x + threadIdx.x; // B*L*DHID
    int    c   = gid & (DHID - 1);
    size_t bt  = gid >> 9;
    int    t   = bt & (SEQ - 1);
    size_t b   = bt >> 12;

    float acc = bias[c];
    for (int j = 0; j < 4; ++j) {
        int d  = 3 - j;
        int tt = rev ? (t + d) : (t - d);
        if (tt >= 0 && tt < SEQ)
            acc += w[c * 4 + j] * src[(b * SEQ + tt) * (size_t)sstride + c];
    }
    float sv = acc / (1.0f + __expf(-acc));
    size_t o = bt * DHID + c;
    if (dst32) dst32[o] = sv;
    dst16[o] = (half_t)sv;
}

// ---------------------------------------------------------------------------
// dt = softplus(x_dbl[:, :32] @ dt_w^T + dt_b)   (K=32: plain VALU, tiny GEMM)
// ---------------------------------------------------------------------------
__global__ void k_dtproj(const float* __restrict__ xdbl,
                         const float* __restrict__ dtw,
                         const float* __restrict__ dtb,
                         float* __restrict__ dt)
{
    size_t gid = (size_t)blockIdx.x * blockDim.x + threadIdx.x; // M*DHID
    size_t row = gid >> 9;
    int    c   = gid & (DHID - 1);
    const float* xr = xdbl + row * 64;
    float acc = dtb[c];
    for (int r = 0; r < RRANK; ++r) acc += xr[r] * dtw[c * RRANK + r];
    dt[gid] = (acc > 20.0f) ? acc : log1pf(__expf(acc));
}

// ---------------------------------------------------------------------------
// Selective scan: one lane per (b,d); 16-state + A in registers; B_t/C_t in
// LDS (shared across the 256 d-channels of the block). rev flips time order.
// ---------------------------------------------------------------------------
__global__ __launch_bounds__(256) void k_scan(
    const float* __restrict__ u,    const float* __restrict__ dt,
    const float* __restrict__ xdbl, const float* __restrict__ Alog,
    const float* __restrict__ Dp,   float* __restrict__ y, int rev)
{
    __shared__ float sBC[32];                  // B_t[0:16], C_t[0:16]
    int b = blockIdx.x >> 1;
    int d = ((blockIdx.x & 1) << 8) + threadIdx.x;   // 0..511

    float A[NSTATE], h[NSTATE];
    for (int n = 0; n < NSTATE; ++n) {
        A[n] = -__expf(Alog[d * NSTATE + n]);
        h[n] = 0.0f;
    }
    float Dv = Dp[d];

    for (int t = 0; t < SEQ; ++t) {
        int tp = rev ? (SEQ - 1 - t) : t;
        size_t row = (size_t)b * SEQ + tp;
        if (threadIdx.x < 32)
            sBC[threadIdx.x] = xdbl[row * 64 + RRANK + threadIdx.x];
        __syncthreads();

        float dtv = dt[row * DHID + d];
        float uv  = u[row * DHID + d];
        float du  = dtv * uv;
        float acc = 0.0f;
        for (int n = 0; n < NSTATE; ++n) {
            h[n] = h[n] * __expf(dtv * A[n]) + du * sBC[n];
            acc += h[n] * sBC[NSTATE + n];
        }
        y[row * DHID + d] = acc + uv * Dv;
        __syncthreads();
    }
}

// ---------------------------------------------------------------------------
// yz16 = (f16)( y * silu(z) ), z = xz[:, 512:1024]
// ---------------------------------------------------------------------------
__global__ void k_yz(const float* __restrict__ y,
                     const float* __restrict__ xz,
                     half_t* __restrict__ yz16)
{
    size_t gid = (size_t)blockIdx.x * blockDim.x + threadIdx.x; // M*DHID
    size_t row = gid >> 9;
    int    c   = gid & (DHID - 1);
    float  z   = xz[row * 1024 + DHID + c];
    float  sv  = z / (1.0f + __expf(-z));
    yz16[gid]  = (half_t)(y[gid] * sv);
}

// ---------------------------------------------------------------------------
// Host orchestration
// ---------------------------------------------------------------------------
extern "C" void kernel_launch(void* const* d_in, const int* in_sizes, int n_in,
                              void* d_out, int out_size, void* d_ws, size_t ws_size,
                              hipStream_t stream)
{
    (void)in_sizes; (void)n_in; (void)out_size; (void)ws_size;

    const float* x      = (const float*)d_in[0];
    const float* ln_g   = (const float*)d_in[1];
    const float* ln_b   = (const float*)d_in[2];
    const float* in_w   = (const float*)d_in[3];
    const float* conv_w = (const float*)d_in[4];
    const float* conv_b = (const float*)d_in[5];
    const float* out_w  = (const float*)d_in[6];
    float* out = (float*)d_out;

    // workspace carve-up (all sizes multiples of 256B)
    char* ws = (char*)d_ws;
    size_t off = 0;
    auto carve = [&](size_t bytes) -> void* {
        void* p = ws + off;
        off += (bytes + 255) & ~(size_t)255;
        return p;
    };
    const size_t M = MROWS;
    half_t* h16      = (half_t*)carve(M * DMODEL * 2);
    float*  proj     = (float*) carve(M * 1024   * 4);   // [conv_in | gate]
    half_t* act16    = (half_t*)carve(M * DHID   * 2);
    float*  xz       = (float*) carve(M * 1024   * 4);   // [x | z]
    float*  xc32     = (float*) carve(M * DHID   * 4);
    half_t* xc16     = (half_t*)carve(M * DHID   * 2);
    float*  xdbl     = (float*) carve(M * 64     * 4);
    float*  dtb      = (float*) carve(M * DHID   * 4);
    float*  y32      = (float*) carve(M * DHID   * 4);
    half_t* yz16     = (half_t*)carve(M * DHID   * 2);
    half_t* gated16  = (half_t*)carve(M * 1024   * 2);
    half_t* inw16    = (half_t*)carve(1024 * 256 * 2);
    half_t* outw16   = (half_t*)carve(256 * 1024 * 2);
    half_t* minw16   = (half_t*)carve(1024 * 512 * 2);
    half_t* mxproj16 = (half_t*)carve(64 * 512   * 2);
    half_t* moutw16  = (half_t*)carve(512 * 512  * 2);

    auto cvt = [&](const float* s, half_t* d, int n) {
        k_f32_to_f16<<<(n + 255) / 256, 256, 0, stream>>>(s, d, n);
    };

    // weights used once
    cvt(in_w,  inw16,  1024 * 256);
    cvt(out_w, outw16, 256 * 1024);

    // 1) LayerNorm -> h16
    k_layernorm<<<M / 8, 256, 0, stream>>>(x, ln_g, ln_b, h16);

    // 2) proj = h @ in_w^T  (M x 1024, K=256)
    k_wmma_gemm<<<dim3(1024 / BN, M / BM), 256, 0, stream>>>(
        h16, DMODEL, inw16, DMODEL, proj, 1024,
        nullptr, 0, 0, nullptr, 0, (int)M, 1024, DMODEL, EPI_F32);

    // 3) act = silu(dwconv(proj[:, :512]))  -> act16 only
    k_dwconv_silu<<<(M * DHID) / 256, 256, 0, stream>>>(
        proj, 1024, conv_w, conv_b, nullptr, act16, 0);

    // 4) fwd / bwd Mamba passes
    for (int pass = 0; pass < 2; ++pass) {
        int base = 7 + pass * 9;
        const float* m_in_w   = (const float*)d_in[base + 0];
        const float* m_conv_w = (const float*)d_in[base + 1];
        const float* m_conv_b = (const float*)d_in[base + 2];
        const float* m_xproj  = (const float*)d_in[base + 3];
        const float* m_dt_w   = (const float*)d_in[base + 4];
        const float* m_dt_b   = (const float*)d_in[base + 5];
        const float* m_Alog   = (const float*)d_in[base + 6];
        const float* m_D      = (const float*)d_in[base + 7];
        const float* m_out_w  = (const float*)d_in[base + 8];
        int rev = pass;  // pass 1 = backward direction

        cvt(m_in_w,  minw16,  1024 * 512);
        cvt(m_xproj, mxproj16, 64 * 512);
        cvt(m_out_w, moutw16, 512 * 512);

        // xz = act @ m_in_w^T  (M x 1024, K=512)
        k_wmma_gemm<<<dim3(1024 / BN, M / BM), 256, 0, stream>>>(
            act16, DHID, minw16, DHID, xz, 1024,
            nullptr, 0, 0, nullptr, 0, (int)M, 1024, DHID, EPI_F32);

        // xc = silu(dwconv(xz[:, :512], direction=rev))
        k_dwconv_silu<<<(M * DHID) / 256, 256, 0, stream>>>(
            xz, 1024, m_conv_w, m_conv_b, xc32, xc16, rev);

        // x_dbl = xc @ xproj^T  (M x 64, K=512)
        k_wmma_gemm<<<dim3(64 / BN, M / BM), 256, 0, stream>>>(
            xc16, DHID, mxproj16, DHID, xdbl, 64,
            nullptr, 0, 0, nullptr, 0, (int)M, 64, DHID, EPI_F32);

        // dt = softplus(x_dbl[:, :32] @ dt_w^T + dt_b)
        k_dtproj<<<(M * DHID) / 256, 256, 0, stream>>>(xdbl, m_dt_w, m_dt_b, dtb);

        // selective scan (sequential over L, direction=rev)
        k_scan<<<BATCH * 2, 256, 0, stream>>>(
            xc32, dtb, xdbl, m_Alog, m_D, y32, rev);

        // yz = y * silu(z)
        k_yz<<<(M * DHID) / 256, 256, 0, stream>>>(y32, xz, yz16);

        // mamba_out = yz @ m_out_w^T; fused: * silu(gate) -> gated16 half
        k_wmma_gemm<<<dim3(512 / BN, M / BM), 256, 0, stream>>>(
            yz16, DHID, moutw16, DHID, nullptr, 0,
            gated16, 1024, pass == 0 ? 0 : 512,
            proj + DHID, 1024, (int)M, DHID, DHID, EPI_GATE16);
    }

    // 5) out = gated @ out_w^T + residual  (M x 256, K=1024)
    k_wmma_gemm<<<dim3(256 / BN, M / BM), 256, 0, stream>>>(
        gated16, 1024, outw16, 1024, out, DMODEL,
        nullptr, 0, 0, x, DMODEL, (int)M, DMODEL, 1024, EPI_RESID);
}
